// StandardAttention_43224550867612
// MI455X (gfx1250) — matrix operand order repaired and checked by reading, not compiled, not run on hardware
//
#include <hip/hip_runtime.h>
#include <math.h>

// ---------------- types ----------------
typedef __attribute__((ext_vector_type(16))) __bf16 v16bf;
typedef __attribute__((ext_vector_type(8)))  float  v8f;
typedef unsigned short bf16s;   // bf16 storage (raw bits) to keep host pass simple

#define B_  2
#define S_  2048
#define D_  1024
#define H_  16
#define HD_ 64
#define NT  (B_ * S_)           // 4096 tokens

// round-to-nearest-even fp32 -> bf16 bits
static __device__ __forceinline__ bf16s f2bf(float f) {
  unsigned u = __builtin_bit_cast(unsigned, f);
  unsigned r = u + 0x7FFFu + ((u >> 16) & 1u);
  return (bf16s)(r >> 16);
}

union FragU { v16bf v; uint4 q[2]; };

// A-matrix fragment (16x32 bf16), src points at (row0, k0) of row-major [ld] matrix.
// Contiguous k order used here: lane<16: M=lane, K=k0..k0+15; lane>=16: same M, K=k0+16..k0+31.
static __device__ __forceinline__ v16bf load_frag_a(const bf16s* src, int ld, int lane) {
  int hi = lane >> 4;
  int m  = lane & 15;
  const bf16s* p = src + m * ld + hi * 16;
  FragU f;
  f.q[0] = *(const uint4*)(p);
  f.q[1] = *(const uint4*)(p + 8);
  return f.v;
}

// B-matrix fragment (32x16 bf16) from N-major source Bt[N][K] (row-major, stride ld).
// Contiguous k order used here: lane<16: N=lane, K=k0..k0+15; lane>=16: N=lane-16, K=k0+16..k0+31.
static __device__ __forceinline__ v16bf load_frag_b(const bf16s* src, int ld, int lane) {
  int hi = lane >> 4;
  int n  = lane & 15;
  const bf16s* p = src + n * ld + hi * 16;
  FragU f;
  f.q[0] = *(const uint4*)(p);
  f.q[1] = *(const uint4*)(p + 8);
  return f.v;
}

static __device__ __forceinline__ v8f wmma_bf16(v16bf a, v16bf b, v8f c) {
  return __builtin_amdgcn_wmma_f32_16x16x32_bf16(false, a, false, b, (short)0, c, false, false);
}

// ---- CDNA5 async global->LDS copy (16 B per lane), tracked by ASYNCcnt ----
// lds_off: wave-relative LDS byte offset (= low 32 bits of the flat shared pointer,
// since the LDS aperture maps addr[31:0] directly to the LDS offset).
static __device__ __forceinline__ void async_copy_b128(unsigned lds_off, const void* gsrc) {
  asm volatile("global_load_async_to_lds_b128 %0, %1, off"
               :: "v"(lds_off), "v"(gsrc)
               : "memory");
}
static __device__ __forceinline__ void wait_async0() {
  asm volatile("s_wait_asynccnt 0x0" ::: "memory");
}

// ---------------- prep kernels ----------------
__global__ void cast_x_kernel(const float* __restrict__ x, bf16s* __restrict__ xb, int n) {
  int i = blockIdx.x * blockDim.x + threadIdx.x;
  if (i < n) xb[i] = f2bf(x[i]);
}

// w [rows][cols] f32  ->  wt [cols][rows] bf16   (so B-frag loads are contiguous)
__global__ void transpose_cast_kernel(const float* __restrict__ w, bf16s* __restrict__ wt,
                                      int rows, int cols) {
  int i = blockIdx.x * blockDim.x + threadIdx.x;
  if (i < rows * cols) {
    int r = i / cols, c = i % cols;
    wt[c * rows + r] = f2bf(w[i]);
  }
}

// ---- shared GEMM main loop: one wave computes a 16(M) x 64(N) tile over K=D_,
// software-pipelined one k-step ahead so loads overlap WMMAs ----
static __device__ __forceinline__ void gemm_16x64(const bf16s* __restrict__ Arow,
                                                  const bf16s* __restrict__ Bt,
                                                  int lane, v8f acc[4]) {
  v16bf a  = load_frag_a(Arow, D_, lane);
  v16bf b0 = load_frag_b(Bt +  0 * D_, D_, lane);
  v16bf b1 = load_frag_b(Bt + 16 * D_, D_, lane);
  v16bf b2 = load_frag_b(Bt + 32 * D_, D_, lane);
  v16bf b3 = load_frag_b(Bt + 48 * D_, D_, lane);
  for (int k0 = 32; k0 < D_; k0 += 32) {
    v16bf an = load_frag_a(Arow + k0, D_, lane);
    v16bf c0 = load_frag_b(Bt +  0 * D_ + k0, D_, lane);
    v16bf c1 = load_frag_b(Bt + 16 * D_ + k0, D_, lane);
    v16bf c2 = load_frag_b(Bt + 32 * D_ + k0, D_, lane);
    v16bf c3 = load_frag_b(Bt + 48 * D_ + k0, D_, lane);
    acc[0] = wmma_bf16(a, b0, acc[0]);
    acc[1] = wmma_bf16(a, b1, acc[1]);
    acc[2] = wmma_bf16(a, b2, acc[2]);
    acc[3] = wmma_bf16(a, b3, acc[3]);
    a = an; b0 = c0; b1 = c1; b2 = c2; b3 = c3;
  }
  acc[0] = wmma_bf16(a, b0, acc[0]);
  acc[1] = wmma_bf16(a, b1, acc[1]);
  acc[2] = wmma_bf16(a, b2, acc[2]);
  acc[3] = wmma_bf16(a, b3, acc[3]);
}

// ---------------- GEMM 1: qkv = X @ Wqkv, scatter to per-head Q/K/V ----------------
// grid (3072/64, 4096/128), block 256 (8 waves). Wave computes a 16(M) x 64(N) tile.
__global__ __launch_bounds__(256)
void gemm_qkv_kernel(const bf16s* __restrict__ X,    // [4096,1024]
                     const bf16s* __restrict__ WT,   // [3072,1024]  (Wqkv^T)
                     bf16s* __restrict__ Q, bf16s* __restrict__ K, bf16s* __restrict__ V) {
  int lane = threadIdx.x & 31;
  int wave = threadIdx.x >> 5;
  int m0 = blockIdx.y * 128 + wave * 16;
  int n0 = blockIdx.x * 64;

  v8f acc[4] = {};
  gemm_16x64(X + (size_t)m0 * D_, WT + (size_t)n0 * D_, lane, acc);

  int rbase = (lane >> 4) * 8;     // C layout: lane<16 rows 0..7, lane>=16 rows 8..15
  int nlane = lane & 15;
#pragma unroll
  for (int t = 0; t < 4; ++t) {
    int n = n0 + t * 16 + nlane;
    int which = n >> 10;           // 0=q 1=k 2=v
    int c = n & 1023;
    int h = c >> 6;
    int d = c & 63;
    bf16s* dst = (which == 0) ? Q : ((which == 1) ? K : V);
#pragma unroll
    for (int r = 0; r < 8; ++r) {
      int m = m0 + rbase + r;          // token index
      int bb = m >> 11;                // / S_
      int ss = m & (S_ - 1);
      dst[(((bb * H_ + h) * S_) + ss) * HD_ + d] = f2bf(acc[t][r]);
    }
  }
}

// ---------------- fused attention (flash-style, online softmax) ----------------
// grid (B*H, S/128), block 256 (8 waves); wave w owns q rows [q0, q0+16).
__global__ __launch_bounds__(256)
void attn_kernel(const bf16s* __restrict__ Q,   // [B,H,S,HD] bf16
                 const bf16s* __restrict__ K,
                 const bf16s* __restrict__ V,
                 bf16s* __restrict__ O) {       // [B*S, D] head-concat bf16
  __shared__ __align__(32) bf16s Kc[32][64];       // K chunk  [key][d]   4 KB
  __shared__ __align__(32) bf16s Vt[64][32];       // V^T chunk [d][key]  4 KB
  __shared__ __align__(32) bf16s Pl[8][16][32];    // per-wave P tiles    8 KB

  int lane = threadIdx.x & 31;
  int wave = threadIdx.x >> 5;
  int bh = blockIdx.x;
  int b = bh >> 4, h = bh & 15;
  int q0 = blockIdx.y * 128 + wave * 16;

  const bf16s* Qp = Q + (size_t)((b * H_ + h) * S_) * HD_;
  const bf16s* Kp = K + (size_t)((b * H_ + h) * S_) * HD_;
  const bf16s* Vp = V + (size_t)((b * H_ + h) * S_) * HD_;

  v16bf qf0 = load_frag_a(Qp + q0 * HD_ + 0,  HD_, lane);
  v16bf qf1 = load_frag_a(Qp + q0 * HD_ + 32, HD_, lane);

  v8f o0 = {}, o1 = {}, o2 = {}, o3 = {};
  float mi[8], li[8];
#pragma unroll
  for (int r = 0; r < 8; ++r) { mi[r] = -__builtin_huge_valf(); li[r] = 0.f; }
  const float scale = 0.125f;   // 1/sqrt(64)

  const unsigned kc_base = (unsigned)(size_t)&Kc[0][0];   // LDS offset of Kc

  for (int j0 = 0; j0 < S_; j0 += 32) {
    __syncthreads();
    // --- async stage K chunk: 32x64 bf16 = 4 KB, contiguous in global and in LDS ---
    {
      unsigned t = threadIdx.x;
      const char* gsrc = (const char*)(Kp + (size_t)j0 * HD_) + t * 16u;
      async_copy_b128(kc_base + t * 16u, gsrc);
    }
    // --- VALU stage of V^T (transpose): thread t -> key row t>>3, 8 d-columns ---
    {
      int kk = threadIdx.x >> 3;
      int d0 = (threadIdx.x & 7) * 8;
      union { uint4 q; bf16s e[8]; } vrow;
      vrow.q = *(const uint4*)(Vp + (size_t)(j0 + kk) * HD_ + d0);
#pragma unroll
      for (int i = 0; i < 8; ++i) Vt[d0 + i][kk] = vrow.e[i];
    }
    wait_async0();          // this wave's async writes are in LDS
    __syncthreads();        // ... and everyone else's

    // scores: S[16q x 32keys] = Q[16x64] @ K^T ; two 16x16 tiles, 2 WMMAs each
    v8f s0 = {}, s1 = {};
    s0 = wmma_bf16(qf0, load_frag_b(&Kc[0][0],   64, lane), s0);
    s0 = wmma_bf16(qf1, load_frag_b(&Kc[0][32],  64, lane), s0);
    s1 = wmma_bf16(qf0, load_frag_b(&Kc[16][0],  64, lane), s1);
    s1 = wmma_bf16(qf1, load_frag_b(&Kc[16][32], 64, lane), s1);

    // online softmax (rows split across 16-lane halves; xor masks 1..8 stay in-half)
    float p0[8], p1[8];
#pragma unroll
    for (int r = 0; r < 8; ++r) {
      float v0 = s0[r] * scale, v1 = s1[r] * scale;
      float cm = fmaxf(v0, v1);
#pragma unroll
      for (int off = 1; off < 16; off <<= 1)
        cm = fmaxf(cm, __shfl_xor(cm, off, 32));
      float nm = fmaxf(mi[r], cm);
      float corr = __expf(mi[r] - nm);
      mi[r] = nm;
      float e0 = __expf(v0 - nm), e1 = __expf(v1 - nm);
      float rs = e0 + e1;
#pragma unroll
      for (int off = 1; off < 16; off <<= 1)
        rs += __shfl_xor(rs, off, 32);
      li[r] = li[r] * corr + rs;
      o0[r] *= corr; o1[r] *= corr; o2[r] *= corr; o3[r] *= corr;
      p0[r] = e0; p1[r] = e1;
    }

    // C-layout -> A-layout for P via per-wave LDS bounce
    {
      int mrow = (lane >> 4) * 8;
      int nc = lane & 15;
#pragma unroll
      for (int r = 0; r < 8; ++r) {
        Pl[wave][mrow + r][nc]      = f2bf(p0[r]);
        Pl[wave][mrow + r][16 + nc] = f2bf(p1[r]);
      }
    }
    v16bf pf = load_frag_a(&Pl[wave][0][0], 32, lane);
    o0 = wmma_bf16(pf, load_frag_b(&Vt[0][0],  32, lane), o0);
    o1 = wmma_bf16(pf, load_frag_b(&Vt[16][0], 32, lane), o1);
    o2 = wmma_bf16(pf, load_frag_b(&Vt[32][0], 32, lane), o2);
    o3 = wmma_bf16(pf, load_frag_b(&Vt[48][0], 32, lane), o3);
  }

  // normalize and store head-concat [token][h*64 + d]
  int mrow = (lane >> 4) * 8;
  int nc = lane & 15;
#pragma unroll
  for (int r = 0; r < 8; ++r) {
    float inv = (li[r] > 0.f) ? 1.f / li[r] : 0.f;
    size_t tok = (size_t)(b * S_ + q0 + mrow + r);
    bf16s* row = O + tok * D_ + h * HD_;
    row[ 0 + nc] = f2bf(o0[r] * inv);
    row[16 + nc] = f2bf(o1[r] * inv);
    row[32 + nc] = f2bf(o2[r] * inv);
    row[48 + nc] = f2bf(o3[r] * inv);
  }
}

// ---------------- GEMM 2: out = A @ Wout (f32 output) ----------------
// grid (1024/64, 4096/128), block 256.
__global__ __launch_bounds__(256)
void gemm_out_kernel(const bf16s* __restrict__ A,    // [4096,1024]
                     const bf16s* __restrict__ WT,   // [1024,1024] (Wout^T)
                     float* __restrict__ out) {
  int lane = threadIdx.x & 31;
  int wave = threadIdx.x >> 5;
  int m0 = blockIdx.y * 128 + wave * 16;
  int n0 = blockIdx.x * 64;

  v8f acc[4] = {};
  gemm_16x64(A + (size_t)m0 * D_, WT + (size_t)n0 * D_, lane, acc);

  int rbase = (lane >> 4) * 8;
  int nlane = lane & 15;
#pragma unroll
  for (int t = 0; t < 4; ++t)
#pragma unroll
    for (int r = 0; r < 8; ++r)
      out[(size_t)(m0 + rbase + r) * D_ + n0 + t * 16 + nlane] = acc[t][r];
}

// ---------------- launcher ----------------
extern "C" void kernel_launch(void* const* d_in, const int* in_sizes, int n_in,
                              void* d_out, int out_size, void* d_ws, size_t ws_size,
                              hipStream_t stream) {
  const float* x    = (const float*)d_in[0];   // [B,S,D]
  const float* Wqkv = (const float*)d_in[1];   // [D,3D]
  const float* Wout = (const float*)d_in[2];   // [D,D]
  float* out = (float*)d_out;

  char* ws = (char*)d_ws;
  size_t off = 0;
  auto salloc = [&](size_t bytes) -> void* {
    void* p = ws + off;
    off = (off + bytes + 255) & ~(size_t)255;
    return p;
  };
  bf16s* Xb    = (bf16s*)salloc((size_t)NT * D_ * 2);
  bf16s* WqkvT = (bf16s*)salloc((size_t)3 * D_ * D_ * 2);
  bf16s* WoutT = (bf16s*)salloc((size_t)D_ * D_ * 2);
  bf16s* Qb    = (bf16s*)salloc((size_t)NT * D_ * 2);
  bf16s* Kb    = (bf16s*)salloc((size_t)NT * D_ * 2);
  bf16s* Vb    = (bf16s*)salloc((size_t)NT * D_ * 2);
  bf16s* Ab    = (bf16s*)salloc((size_t)NT * D_ * 2);

  cast_x_kernel<<<(NT * D_ + 255) / 256, 256, 0, stream>>>(x, Xb, NT * D_);
  transpose_cast_kernel<<<(D_ * 3 * D_ + 255) / 256, 256, 0, stream>>>(Wqkv, WqkvT, D_, 3 * D_);
  transpose_cast_kernel<<<(D_ * D_ + 255) / 256, 256, 0, stream>>>(Wout, WoutT, D_, D_);

  gemm_qkv_kernel<<<dim3(3 * D_ / 64, NT / 128), 256, 0, stream>>>(Xb, WqkvT, Qb, Kb, Vb);
  attn_kernel<<<dim3(B_ * H_, S_ / 128), 256, 0, stream>>>(Qb, Kb, Vb, Ab);
  gemm_out_kernel<<<dim3(D_ / 64, NT / 128), 256, 0, stream>>>(Ab, WoutT, out);
}
